// InvertedAttentionBlock_46308337385576
// MI455X (gfx1250) — matrix operand order repaired and checked by reading, not compiled, not run on hardware
//
#include <hip/hip_runtime.h>
#include <hip/hip_bf16.h>
#include <math.h>

#define S_DIM 1024
#define B_DIM 8
#define E_DIM 768
#define H_DIM 12
#define D_DIM 64
#define HID_DIM 3072
#define NROWS (S_DIM * B_DIM) /* 8192 */

typedef _Float16 half_t;
typedef __attribute__((ext_vector_type(16))) _Float16 v16h;
typedef __attribute__((ext_vector_type(8)))  _Float16 v8h;
typedef __attribute__((ext_vector_type(8)))  float    v8f;

static __device__ __forceinline__ v16h cat16(v8h lo, v8h hi) {
  return __builtin_shufflevector(lo, hi, 0, 1, 2, 3, 4, 5, 6, 7, 8, 9, 10, 11, 12, 13, 14, 15);
}

// A fragment (16x32 f16). `A` points at element (tileRow0, k0) of a row-major
// matrix with leading dimension `ld` (halves). ISA 16-bit A layout:
// lane holds M = lane&15; halves j: K = (j/8)*16 + (lane>>4)*8 + (j%8).
static __device__ __forceinline__ v16h load_a_frag(const half_t* A, int ld, int lane) {
  const half_t* p = A + (size_t)(lane & 15) * ld + ((lane >> 4) << 3);
  v8h lo = *reinterpret_cast<const v8h*>(p);
  v8h hi = *reinterpret_cast<const v8h*>(p + 16);
  return cat16(lo, hi);
}

// B fragment (32x16 f16). `Wp` points at element (col0, k0) of a row-major
// [N,K] matrix with leading dimension `ld`. ISA B layout: lane holds
// N = lane&15; halves j: K = (j/8)*16 + (lane>>4)*8 + (j%8).
static __device__ __forceinline__ v16h load_b_frag(const half_t* Wp, int ld, int lane) {
  const half_t* p = Wp + (size_t)(lane & 15) * ld + ((lane >> 4) << 3);
  v8h lo = *reinterpret_cast<const v8h*>(p);
  v8h hi = *reinterpret_cast<const v8h*>(p + 16);
  return cat16(lo, hi);
}

static __device__ __forceinline__ v8f wmma_f16(v16h a, v16h b, v8f c) {
  return __builtin_amdgcn_wmma_f32_16x16x32_f16(false, a, false, b, (short)0, c, false, false);
}

// ---------------------------------------------------------------------------
// Per-wave GEMM core, ping-pong pipelined: C(32x64) += A(32xK) * W(64xK)^T.
// Two named buffer sets (X/Y) refilled by loads only -> no rotation movs.
// Requires K % 64 == 0 (the +64 refill is offset-clamped on the last iter).
// ---------------------------------------------------------------------------
template <int K>
static __device__ __forceinline__ void gemm32x64(const half_t* __restrict__ A,
                                                 const half_t* __restrict__ W,
                                                 int rowBase, int colBase, int lane,
                                                 v8f acc[2][4]) {
  const half_t* a0 = A + (size_t)rowBase * K;
  const half_t* a1 = a0 + (size_t)16 * K;
  const half_t* w0 = W + (size_t)colBase * K;

  v16h ax0 = load_a_frag(a0, K, lane);
  v16h ax1 = load_a_frag(a1, K, lane);
  v16h bx0 = load_b_frag(w0, K, lane);
  v16h bx1 = load_b_frag(w0 + (size_t)16 * K, K, lane);
  v16h bx2 = load_b_frag(w0 + (size_t)32 * K, K, lane);
  v16h bx3 = load_b_frag(w0 + (size_t)48 * K, K, lane);

#pragma unroll 1
  for (int k0 = 0; k0 < K; k0 += 64) {
    int k1 = k0 + 32; // always < K since K % 64 == 0
    v16h ay0 = load_a_frag(a0 + k1, K, lane);
    v16h ay1 = load_a_frag(a1 + k1, K, lane);
    v16h by0 = load_b_frag(w0 + k1, K, lane);
    v16h by1 = load_b_frag(w0 + (size_t)16 * K + k1, K, lane);
    v16h by2 = load_b_frag(w0 + (size_t)32 * K + k1, K, lane);
    v16h by3 = load_b_frag(w0 + (size_t)48 * K + k1, K, lane);

    acc[0][0] = wmma_f16(ax0, bx0, acc[0][0]);
    acc[0][1] = wmma_f16(ax0, bx1, acc[0][1]);
    acc[0][2] = wmma_f16(ax0, bx2, acc[0][2]);
    acc[0][3] = wmma_f16(ax0, bx3, acc[0][3]);
    acc[1][0] = wmma_f16(ax1, bx0, acc[1][0]);
    acc[1][1] = wmma_f16(ax1, bx1, acc[1][1]);
    acc[1][2] = wmma_f16(ax1, bx2, acc[1][2]);
    acc[1][3] = wmma_f16(ax1, bx3, acc[1][3]);

    int k2 = (k0 + 64 < K) ? (k0 + 64) : 0; // clamp: harmless redundant refill
    ax0 = load_a_frag(a0 + k2, K, lane);
    ax1 = load_a_frag(a1 + k2, K, lane);
    bx0 = load_b_frag(w0 + k2, K, lane);
    bx1 = load_b_frag(w0 + (size_t)16 * K + k2, K, lane);
    bx2 = load_b_frag(w0 + (size_t)32 * K + k2, K, lane);
    bx3 = load_b_frag(w0 + (size_t)48 * K + k2, K, lane);

    acc[0][0] = wmma_f16(ay0, by0, acc[0][0]);
    acc[0][1] = wmma_f16(ay0, by1, acc[0][1]);
    acc[0][2] = wmma_f16(ay0, by2, acc[0][2]);
    acc[0][3] = wmma_f16(ay0, by3, acc[0][3]);
    acc[1][0] = wmma_f16(ay1, by0, acc[1][0]);
    acc[1][1] = wmma_f16(ay1, by1, acc[1][1]);
    acc[1][2] = wmma_f16(ay1, by2, acc[1][2]);
    acc[1][3] = wmma_f16(ay1, by3, acc[1][3]);
  }
}

// WG = 8 waves: 2 M-subtiles x 4 N-subtiles -> WG tile 64M x 256N.
static __device__ __forceinline__ void tile_coords(int tidx, int bx, int by,
                                                   int& rowBase, int& colBase,
                                                   int& lane) {
  lane = tidx & 31;
  int wave = tidx >> 5;
  rowBase = bx * 64 + (wave & 1) * 32;
  colBase = by * 256 + (wave >> 1) * 64;
}

// ---------------------------------------------------------------------------
// float -> f16 conversion
// ---------------------------------------------------------------------------
__global__ void k_cvt(const float* __restrict__ in, half_t* __restrict__ out, int n) {
  int i = blockIdx.x * blockDim.x + threadIdx.x;
  if (i < n) out[i] = (half_t)in[i];
}

// ---------------------------------------------------------------------------
// LayerNorm over rows of length 768, emits f16. 256 threads, 3 elems/thread.
// ---------------------------------------------------------------------------
__global__ __launch_bounds__(256) void k_ln(const float* __restrict__ xin,
                                            const float* __restrict__ g,
                                            const float* __restrict__ bb,
                                            half_t* __restrict__ out16) {
  __shared__ float sm[8];
  int row = blockIdx.x;
  int t = threadIdx.x;
  const float* xr = xin + (size_t)row * E_DIM;
  float v0 = xr[t], v1 = xr[t + 256], v2 = xr[t + 512];
  float s = v0 + v1 + v2;
#pragma unroll
  for (int off = 16; off >= 1; off >>= 1) s += __shfl_xor(s, off, 32);
  if ((t & 31) == 0) sm[t >> 5] = s;
  __syncthreads();
  float tot = 0.f;
#pragma unroll
  for (int i = 0; i < 8; ++i) tot += sm[i];
  float mu = tot * (1.0f / E_DIM);
  float d0 = v0 - mu, d1 = v1 - mu, d2 = v2 - mu;
  float q = d0 * d0 + d1 * d1 + d2 * d2;
  __syncthreads();
#pragma unroll
  for (int off = 16; off >= 1; off >>= 1) q += __shfl_xor(q, off, 32);
  if ((t & 31) == 0) sm[t >> 5] = q;
  __syncthreads();
  float vtot = 0.f;
#pragma unroll
  for (int i = 0; i < 8; ++i) vtot += sm[i];
  float rstd = rsqrtf(vtot * (1.0f / E_DIM) + 1e-5f);
  half_t* o = out16 + (size_t)row * E_DIM;
  o[t]       = (half_t)(d0 * rstd * g[t] + bb[t]);
  o[t + 256] = (half_t)(d1 * rstd * g[t + 256] + bb[t + 256]);
  o[t + 512] = (half_t)(d2 * rstd * g[t + 512] + bb[t + 512]);
}

// ---------------------------------------------------------------------------
// QKV projection: [8192,768] x [2304,768]^T, scatter into per-head q/k/vT.
// q is pre-scaled by 1/sqrt(D) = 0.125 (exact in f16).
// ---------------------------------------------------------------------------
__global__ __launch_bounds__(256) void k_qkv(const half_t* __restrict__ inp16,
                                             const half_t* __restrict__ w16,
                                             const float* __restrict__ bias,
                                             half_t* __restrict__ q16,
                                             half_t* __restrict__ k16,
                                             half_t* __restrict__ vT16) {
  int rowBase, colBase, lane;
  tile_coords(threadIdx.x, blockIdx.x, blockIdx.y, rowBase, colBase, lane);
  v8f zero = {};
  v8f acc[2][4] = {{zero, zero, zero, zero}, {zero, zero, zero, zero}};
  gemm32x64<E_DIM>(inp16, w16, rowBase, colBase, lane, acc);
  int mgrp = (lane >> 4) << 3, nlan = lane & 15;
#pragma unroll
  for (int mi = 0; mi < 2; ++mi) {
#pragma unroll
    for (int nt = 0; nt < 4; ++nt) {
#pragma unroll
      for (int i = 0; i < 8; ++i) {
        int row = rowBase + mi * 16 + mgrp + i;
        int col = colBase + nt * 16 + nlan;
        float val = acc[mi][nt][i] + bias[col];
        int s = row >> 3, bi = row & 7;
        int which = col / E_DIM;
        int rem = col - which * E_DIM;
        int hh = rem >> 6, dd = rem & 63;
        size_t head = (size_t)(bi * H_DIM + hh) * S_DIM;
        if (which == 0)
          q16[(head + s) * D_DIM + dd] = (half_t)(val * 0.125f);
        else if (which == 1)
          k16[(head + s) * D_DIM + dd] = (half_t)val;
        else
          vT16[((size_t)(bi * H_DIM + hh) * D_DIM + dd) * S_DIM + s] = (half_t)val;
      }
    }
  }
}

// ---------------------------------------------------------------------------
// Flash attention with inverted local-window mask. Block = 4 waves, each wave
// owns 16 query rows; 32 keys per step (4 score WMMAs + 4 ctx WMMAs).
// Ping-pong: two k-fragment buffer sets swapped by role per 64-key iteration;
// next tile's K-fragments and this tile's V-fragments are issued before the
// softmax VALU block so global loads overlap the exp/shuffle work.
// ---------------------------------------------------------------------------
__global__ __launch_bounds__(128) void k_attn(const half_t* __restrict__ q16,
                                              const half_t* __restrict__ k16,
                                              const half_t* __restrict__ vT16,
                                              half_t* __restrict__ ctx16) {
  __shared__ half_t plds[4][16 * 32];
  int lane = threadIdx.x & 31;
  int wave = threadIdx.x >> 5;
  int h = blockIdx.y, b = blockIdx.z;
  int qbase = blockIdx.x * 64 + wave * 16;
  size_t headOff = (size_t)(b * H_DIM + h) * S_DIM * D_DIM;
  const half_t* qp = q16 + headOff;
  const half_t* kp = k16 + headOff;
  const half_t* vp = vT16 + headOff; // [64][S]

  v16h qA0 = load_a_frag(qp + (size_t)qbase * D_DIM, D_DIM, lane);      // d 0..31
  v16h qA1 = load_a_frag(qp + (size_t)qbase * D_DIM + 32, D_DIM, lane); // d 32..63

  v8f zero = {};
  v8f ctx[4] = {zero, zero, zero, zero};
  float rmax[8], rsum[8];
#pragma unroll
  for (int i = 0; i < 8; ++i) { rmax[i] = -1e30f; rsum[i] = 0.f; }

  int mgrp = (lane >> 4) << 3;
  int nlan = lane & 15;

  // One 32-key step: consume passed K-fragments, refill `n*` with tile kb+32.
  auto step = [&](int kb, v16h kf0, v16h kf1, v16h kf2, v16h kf3,
                  v16h& n0, v16h& n1, v16h& n2, v16h& n3) {
    v8f s0 = {}, s1 = {};
    s0 = wmma_f16(qA0, kf0, s0);
    s0 = wmma_f16(qA1, kf1, s0);
    s1 = wmma_f16(qA0, kf2, s1);
    s1 = wmma_f16(qA1, kf3, s1);

    int kn = kb + 32;
    if (kn >= S_DIM) kn = 0; // clamp: harmless redundant refill on last step
    n0 = load_b_frag(kp + (size_t)kn * D_DIM, D_DIM, lane);
    n1 = load_b_frag(kp + (size_t)kn * D_DIM + 32, D_DIM, lane);
    n2 = load_b_frag(kp + (size_t)(kn + 16) * D_DIM, D_DIM, lane);
    n3 = load_b_frag(kp + (size_t)(kn + 16) * D_DIM + 32, D_DIM, lane);
    v16h vB0 = load_b_frag(vp + kb, S_DIM, lane);
    v16h vB1 = load_b_frag(vp + (size_t)16 * S_DIM + kb, S_DIM, lane);
    v16h vB2 = load_b_frag(vp + (size_t)32 * S_DIM + kb, S_DIM, lane);
    v16h vB3 = load_b_frag(vp + (size_t)48 * S_DIM + kb, S_DIM, lane);

    int key0 = kb + nlan, key1 = kb + 16 + nlan;
    int kr0 = key0 >> 5, kc0 = key0 & 31;
    int kr1 = key1 >> 5, kc1 = key1 & 31;

#pragma unroll
    for (int i = 0; i < 8; ++i) {
      int qrow = qbase + mgrp + i;
      int qr = qrow >> 5, qc = qrow & 31;
      float a = s0[i], c = s1[i];
      int dr0 = qr - kr0; dr0 = dr0 < 0 ? -dr0 : dr0;
      int dc0 = qc - kc0; dc0 = dc0 < 0 ? -dc0 : dc0;
      int dr1 = qr - kr1; dr1 = dr1 < 0 ? -dr1 : dr1;
      int dc1 = qc - kc1; dc1 = dc1 < 0 ? -dc1 : dc1;
      if (dr0 <= 4 && dc0 <= 4) a = -1e30f;
      if (dr1 <= 4 && dc1 <= 4) c = -1e30f;
      float t = fmaxf(a, c);
#pragma unroll
      for (int off = 1; off < 16; off <<= 1) t = fmaxf(t, __shfl_xor(t, off, 32));
      float newm = fmaxf(rmax[i], t);
      float alpha = __expf(rmax[i] - newm);
      rmax[i] = newm;
      float p0 = (a < -1e29f) ? 0.f : __expf(a - newm);
      float p1 = (c < -1e29f) ? 0.f : __expf(c - newm);
      float ps = p0 + p1;
#pragma unroll
      for (int off = 1; off < 16; off <<= 1) ps += __shfl_xor(ps, off, 32);
      rsum[i] = rsum[i] * alpha + ps;
#pragma unroll
      for (int d = 0; d < 4; ++d) ctx[d][i] = ctx[d][i] * alpha;
      int m = mgrp + i;
      plds[wave][m * 32 + nlan] = (half_t)p0;
      plds[wave][m * 32 + 16 + nlan] = (half_t)p1;
    }

    // Re-layout P (C-layout) into an A-fragment via this wave's private LDS.
    v16h pA = load_a_frag(&plds[wave][0], 32, lane);
    ctx[0] = wmma_f16(pA, vB0, ctx[0]);
    ctx[1] = wmma_f16(pA, vB1, ctx[1]);
    ctx[2] = wmma_f16(pA, vB2, ctx[2]);
    ctx[3] = wmma_f16(pA, vB3, ctx[3]);
  };

  // prologue: key fragments for tile 0 into set C
  v16h kc0 = load_b_frag(kp, D_DIM, lane);
  v16h kc1 = load_b_frag(kp + 32, D_DIM, lane);
  v16h kc2 = load_b_frag(kp + (size_t)16 * D_DIM, D_DIM, lane);
  v16h kc3 = load_b_frag(kp + (size_t)16 * D_DIM + 32, D_DIM, lane);
  v16h kd0, kd1, kd2, kd3;

#pragma unroll 1
  for (int kb = 0; kb < S_DIM; kb += 64) {
    step(kb, kc0, kc1, kc2, kc3, kd0, kd1, kd2, kd3);
    step(kb + 32, kd0, kd1, kd2, kd3, kc0, kc1, kc2, kc3);
  }

#pragma unroll
  for (int d = 0; d < 4; ++d) {
#pragma unroll
    for (int i = 0; i < 8; ++i) {
      int srow = qbase + mgrp + i;
      int col = h * D_DIM + d * 16 + nlan;
      float val = ctx[d][i] / rsum[i];
      ctx16[((size_t)srow * B_DIM + b) * E_DIM + col] = (half_t)val;
    }
  }
}

// ---------------------------------------------------------------------------
// Out-projection + residual: x2 = x + ctx @ Wo^T + bo   (f32 out)
// ---------------------------------------------------------------------------
__global__ __launch_bounds__(256) void k_outproj(const half_t* __restrict__ ctx16,
                                                 const half_t* __restrict__ wo16,
                                                 const float* __restrict__ bias,
                                                 const float* __restrict__ x,
                                                 float* __restrict__ x2) {
  int rowBase, colBase, lane;
  tile_coords(threadIdx.x, blockIdx.x, blockIdx.y, rowBase, colBase, lane);
  v8f zero = {};
  v8f acc[2][4] = {{zero, zero, zero, zero}, {zero, zero, zero, zero}};
  gemm32x64<E_DIM>(ctx16, wo16, rowBase, colBase, lane, acc);
  int mgrp = (lane >> 4) << 3, nlan = lane & 15;
#pragma unroll
  for (int mi = 0; mi < 2; ++mi) {
#pragma unroll
    for (int nt = 0; nt < 4; ++nt) {
#pragma unroll
      for (int i = 0; i < 8; ++i) {
        size_t row = rowBase + mi * 16 + mgrp + i;
        int col = colBase + nt * 16 + nlan;
        size_t idx = row * E_DIM + col;
        x2[idx] = x[idx] + acc[mi][nt][i] + bias[col];
      }
    }
  }
}

// ---------------------------------------------------------------------------
// MLP layer 1 + exact GELU: hid = gelu(h @ W1^T + b1)  (f16 out)
// ---------------------------------------------------------------------------
__global__ __launch_bounds__(256) void k_mlp1(const half_t* __restrict__ h16,
                                              const half_t* __restrict__ w116,
                                              const float* __restrict__ bias,
                                              half_t* __restrict__ hid16) {
  int rowBase, colBase, lane;
  tile_coords(threadIdx.x, blockIdx.x, blockIdx.y, rowBase, colBase, lane);
  v8f zero = {};
  v8f acc[2][4] = {{zero, zero, zero, zero}, {zero, zero, zero, zero}};
  gemm32x64<E_DIM>(h16, w116, rowBase, colBase, lane, acc);
  int mgrp = (lane >> 4) << 3, nlan = lane & 15;
#pragma unroll
  for (int mi = 0; mi < 2; ++mi) {
#pragma unroll
    for (int nt = 0; nt < 4; ++nt) {
#pragma unroll
      for (int i = 0; i < 8; ++i) {
        size_t row = rowBase + mi * 16 + mgrp + i;
        int col = colBase + nt * 16 + nlan;
        float v = acc[mi][nt][i] + bias[col];
        float ge = 0.5f * v * (1.0f + erff(v * 0.70710678118654752f));
        hid16[row * HID_DIM + col] = (half_t)ge;
      }
    }
  }
}

// ---------------------------------------------------------------------------
// MLP layer 2 + residual: out = x2 + hid @ W2^T + b2   (f32 out, K=3072)
// ---------------------------------------------------------------------------
__global__ __launch_bounds__(256) void k_mlp2(const half_t* __restrict__ hid16,
                                              const half_t* __restrict__ w216,
                                              const float* __restrict__ bias,
                                              const float* __restrict__ x2,
                                              float* __restrict__ out) {
  int rowBase, colBase, lane;
  tile_coords(threadIdx.x, blockIdx.x, blockIdx.y, rowBase, colBase, lane);
  v8f zero = {};
  v8f acc[2][4] = {{zero, zero, zero, zero}, {zero, zero, zero, zero}};
  gemm32x64<HID_DIM>(hid16, w216, rowBase, colBase, lane, acc);
  int mgrp = (lane >> 4) << 3, nlan = lane & 15;
#pragma unroll
  for (int mi = 0; mi < 2; ++mi) {
#pragma unroll
    for (int nt = 0; nt < 4; ++nt) {
#pragma unroll
      for (int i = 0; i < 8; ++i) {
        size_t row = rowBase + mi * 16 + mgrp + i;
        int col = colBase + nt * 16 + nlan;
        size_t idx = row * E_DIM + col;
        out[idx] = x2[idx] + acc[mi][nt][i] + bias[col];
      }
    }
  }
}

// ---------------------------------------------------------------------------
extern "C" void kernel_launch(void* const* d_in, const int* in_sizes, int n_in,
                              void* d_out, int out_size, void* d_ws, size_t ws_size,
                              hipStream_t stream) {
  const float* x     = (const float*)d_in[0];
  const float* ln1_g = (const float*)d_in[1];
  const float* ln1_b = (const float*)d_in[2];
  const float* in_w  = (const float*)d_in[3];
  const float* in_b  = (const float*)d_in[4];
  const float* out_w = (const float*)d_in[5];
  const float* out_b = (const float*)d_in[6];
  const float* ln2_g = (const float*)d_in[7];
  const float* ln2_b = (const float*)d_in[8];
  const float* w1    = (const float*)d_in[9];
  const float* b1    = (const float*)d_in[10];
  const float* w2    = (const float*)d_in[11];
  const float* b2    = (const float*)d_in[12];
  float* out = (float*)d_out;

  char* ws = (char*)d_ws;
  size_t off = 0;
  auto alloc = [&](size_t bytes) -> void* {
    off = (off + 255) & ~(size_t)255;
    void* p = ws + off;
    off += bytes;
    return p;
  };

  const size_t headElems = (size_t)B_DIM * H_DIM * S_DIM * D_DIM; // 6291456
  half_t* w_in16 = (half_t*)alloc((size_t)3 * E_DIM * E_DIM * sizeof(half_t));
  half_t* w_o16  = (half_t*)alloc((size_t)E_DIM * E_DIM * sizeof(half_t));
  half_t* w1_16  = (half_t*)alloc((size_t)HID_DIM * E_DIM * sizeof(half_t));
  half_t* w2_16  = (half_t*)alloc((size_t)E_DIM * HID_DIM * sizeof(half_t));
  half_t* inp16  = (half_t*)alloc((size_t)NROWS * E_DIM * sizeof(half_t));
  half_t* q16    = (half_t*)alloc(headElems * sizeof(half_t));
  half_t* k16    = (half_t*)alloc(headElems * sizeof(half_t));
  half_t* vT16   = (half_t*)alloc(headElems * sizeof(half_t));
  half_t* ctx16  = (half_t*)alloc((size_t)NROWS * E_DIM * sizeof(half_t));
  float*  x2     = (float*)alloc((size_t)NROWS * E_DIM * sizeof(float));
  half_t* h16    = (half_t*)alloc((size_t)NROWS * E_DIM * sizeof(half_t));
  half_t* hid16  = (half_t*)alloc((size_t)NROWS * HID_DIM * sizeof(half_t));

  auto cvt = [&](const float* src, half_t* dst, int n) {
    k_cvt<<<(n + 255) / 256, 256, 0, stream>>>(src, dst, n);
  };
  cvt(in_w, w_in16, 3 * E_DIM * E_DIM);
  cvt(out_w, w_o16, E_DIM * E_DIM);
  cvt(w1, w1_16, HID_DIM * E_DIM);
  cvt(w2, w2_16, E_DIM * HID_DIM);

  // LN1
  k_ln<<<NROWS, 256, 0, stream>>>(x, ln1_g, ln1_b, inp16);
  // QKV projection  (M=8192, N=2304), WG tile 64x256
  k_qkv<<<dim3(NROWS / 64, (3 * E_DIM) / 256), 256, 0, stream>>>(inp16, w_in16, in_b, q16, k16, vT16);
  // Attention (flash, streaming softmax)
  k_attn<<<dim3(S_DIM / 64, H_DIM, B_DIM), 128, 0, stream>>>(q16, k16, vT16, ctx16);
  // Out projection + residual
  k_outproj<<<dim3(NROWS / 64, E_DIM / 256), 256, 0, stream>>>(ctx16, w_o16, out_b, x, x2);
  // LN2
  k_ln<<<NROWS, 256, 0, stream>>>(x2, ln2_g, ln2_b, h16);
  // MLP
  k_mlp1<<<dim3(NROWS / 64, HID_DIM / 256), 256, 0, stream>>>(h16, w1_16, b1, hid16);
  k_mlp2<<<dim3(NROWS / 64, E_DIM / 256), 256, 0, stream>>>(hid16, w2_16, b2, x2, out);

  (void)in_sizes; (void)n_in; (void)out_size; (void)ws_size;
}
